// TwoBranchGNN_34437047780017
// MI455X (gfx1250) — compile-verified, matrix operand order checked by reading, not run on hardware
//
#include <hip/hip_runtime.h>
#include <hip/hip_bf16.h>
#include <stdint.h>

// ---------------------------------------------------------------------------
// Two-branch GCN on MI455X (gfx1250, wave32).
// Dominant cost: 12x (4096x4096) @ (4096xd) aggregations -> bf16 WMMA,
// fp32 accumulate. An (32 MB bf16) fits in the 192 MB L2, so aggregations are
// L2-resident after build_an. A-tiles are staged with async global->LDS
// (ASYNCcnt) to avoid the VGPR round-trip; operands are zero-padded to tile
// multiples so the GEMM has no bounds checks anywhere.
// ---------------------------------------------------------------------------

typedef unsigned short u16;
typedef __attribute__((ext_vector_type(16))) __bf16 v16bf;
typedef __attribute__((ext_vector_type(8)))  float  v8f;
typedef __attribute__((ext_vector_type(4)))  unsigned int v4u;

union FragAB { v16bf v; v4u q[2]; };

__device__ __forceinline__ u16 f2bf_bits(float f) {
  unsigned int u = __float_as_uint(f);
  unsigned int r = u + 0x7FFFu + ((u >> 16) & 1u);   // round-to-nearest-even
  return (u16)(r >> 16);
}

__device__ __forceinline__ unsigned lds_off32(const void* p) {
  // generic pointer to LDS: low 32 bits are the LDS byte offset
  return (unsigned)(uintptr_t)p;
}

enum { MODE_RAW = 0, MODE_BIAS = 1, MODE_BIAS_RELU = 2, MODE_RES_RELU = 3, MODE_RES_LIN = 4 };

#define BM 128
#define BN 64
#define BK 64

// C[MxN] = A[MxK](bf16) @ B[KxN](bf16), fp32 accum, fused epilogue.
// Preconditions (guaranteed by host-side padding): M%128==0, N%64==0, K%64==0,
// all pointers 16B aligned, leading dims == N / K exactly.
template <int MODE>
__global__ __launch_bounds__(256)
void gemm_bf16_wmma(const u16* __restrict__ A,
                    const u16* __restrict__ B,
                    const float* __restrict__ bias,
                    const float* __restrict__ res,
                    float* __restrict__ outF,
                    u16* __restrict__ outB,
                    int M, int N, int K, float scale)
{
  __shared__ u16 As[BM][BK];   // 16 KB
  __shared__ u16 Bs[BN][BK];   // 8 KB, stored transposed (N-major)

  const int tid   = threadIdx.x;
  const int lane  = tid & 31;
  const int wave  = tid >> 5;
  const int waveM = wave >> 1;            // 0..3
  const int waveN = wave & 1;             // 0..1
  const int m0 = blockIdx.y * BM;
  const int n0 = blockIdx.x * BN;

  const int rl = lane & 15;               // row/col inside fragment
  const int kb = (lane >> 4) * 8;         // K sub-block per lane half
  const int rh = (lane >> 4) * 8;         // C-layout row offset

  v8f acc[2][2];
  #pragma unroll
  for (int i = 0; i < 2; ++i)
    #pragma unroll
    for (int j = 0; j < 2; ++j)
      #pragma unroll
      for (int e = 0; e < 8; ++e) acc[i][j][e] = 0.0f;

  // A-tile loader: thread -> (row 0..127, 32-col half); 4 async b128 per thread
  const int arow = tid >> 1;
  const int acol = (tid & 1) * 32;
  // B-tile loader: thread -> (k-row 0..63, 16-col chunk); 2 x b128 per thread
  const int brow = tid >> 2;
  const int bcol = (tid & 3) * 16;

  const u16* Arow  = A + (size_t)(m0 + arow) * K + acol;
  const u16* Brow0 = B + (size_t)brow * N + (n0 + bcol);
  const unsigned ldsA = lds_off32(&As[arow][acol]);

  for (int k0 = 0; k0 < K; k0 += BK) {
    // ---- stage A tile: async global->LDS (no VGPR round-trip, ASYNCcnt) ----
    {
      const unsigned long long ga = (unsigned long long)(uintptr_t)(Arow + k0);
      #pragma unroll
      for (int e = 0; e < 4; ++e) {
        asm volatile("global_load_async_to_lds_b128 %0, %1, off offset:%2"
                     :: "v"(ldsA), "v"(ga), "i"(e * 16)
                     : "memory");
      }
    }
    // ---- stage B tile, transposed into Bs[n][k] ----
    {
      const u16* sb = Brow0 + (size_t)k0 * N;
      u16 tmp[16];
      *(v4u*)&tmp[0] = *(const v4u*)sb;
      *(v4u*)&tmp[8] = *(const v4u*)(sb + 8);
      #pragma unroll
      for (int e = 0; e < 16; ++e) Bs[bcol + e][brow] = tmp[e];
    }
    // fence the async A-tile writes, then rendezvous
    asm volatile("s_wait_asynccnt 0x0" ::: "memory");
    __syncthreads();

    // ---- 2 WMMA K-steps per stage ----
    #pragma unroll
    for (int kk = 0; kk < BK; kk += 32) {
      FragAB afr[2], bfr[2];
      #pragma unroll
      for (int f = 0; f < 2; ++f) {
        const int mrow = waveM * 32 + f * 16 + rl;
        afr[f].q[0] = *(const v4u*)&As[mrow][kk + kb];
        afr[f].q[1] = *(const v4u*)&As[mrow][kk + kb + 16];
        const int ncol = waveN * 32 + f * 16 + rl;
        bfr[f].q[0] = *(const v4u*)&Bs[ncol][kk + kb];
        bfr[f].q[1] = *(const v4u*)&Bs[ncol][kk + kb + 16];
      }
      #pragma unroll
      for (int fm = 0; fm < 2; ++fm)
        #pragma unroll
        for (int fn = 0; fn < 2; ++fn)
          acc[fm][fn] = __builtin_amdgcn_wmma_f32_16x16x32_bf16(
              false, afr[fm].v, false, bfr[fn].v,
              (short)0, acc[fm][fn], false, false);
    }
    __syncthreads();
  }

  // ---- fused epilogue (no bounds checks; padding lanes carry exact zeros) ----
  #pragma unroll
  for (int fm = 0; fm < 2; ++fm) {
    #pragma unroll
    for (int fn = 0; fn < 2; ++fn) {
      const int col = n0 + waveN * 32 + fn * 16 + rl;
      const float b = (MODE != MODE_RAW) ? bias[col] : 0.0f;
      #pragma unroll
      for (int r = 0; r < 8; ++r) {
        const int row = m0 + waveM * 32 + fm * 16 + rh + r;
        const size_t idx = (size_t)row * N + col;
        float v = acc[fm][fn][r] + b;
        if (MODE == MODE_BIAS_RELU)     v = fmaxf(v, 0.0f);
        if (MODE == MODE_RES_RELU)      v = res[idx] + scale * fmaxf(v, 0.0f);
        if (MODE == MODE_RES_LIN)       v = res[idx] + scale * v;
        if (outF) outF[idx] = v;
        if (outB) outB[idx] = f2bf_bits(v);
      }
    }
  }
}

// dis[row] = rsqrt(1 + sum_j A[row][j])   (self-loop folded in)
__global__ __launch_bounds__(256)
void rowsum_rsqrt_kernel(const float* __restrict__ A, float* __restrict__ dis, int n)
{
  __shared__ float red[256];
  const int row = blockIdx.x;
  float s = 0.0f;
  for (int j = threadIdx.x; j < n; j += 256) s += A[(size_t)row * n + j];
  red[threadIdx.x] = s;
  __syncthreads();
  for (int st = 128; st > 0; st >>= 1) {
    if (threadIdx.x < st) red[threadIdx.x] += red[threadIdx.x + st];
    __syncthreads();
  }
  if (threadIdx.x == 0) dis[row] = rsqrtf(red[0] + 1.0f);
}

// an[i][j] = bf16((A[i][j] + (i==j)) * dis[i] * dis[j])
__global__ __launch_bounds__(256)
void build_an_kernel(const float* __restrict__ A, const float* __restrict__ dis,
                     u16* __restrict__ an, int n)
{
  const size_t total = (size_t)n * n;
  const size_t stride = (size_t)gridDim.x * blockDim.x;
  for (size_t i = (size_t)blockIdx.x * blockDim.x + threadIdx.x; i < total; i += stride) {
    const int r = (int)(i / n);
    const int c = (int)(i % n);
    float v = A[i] + (r == c ? 1.0f : 0.0f);
    an[i] = f2bf_bits(v * dis[r] * dis[c]);
  }
}

// dst[dr x dc] (bf16) = zero-padded convert of src[sr x sc] (f32)
__global__ __launch_bounds__(256)
void cvt_pad2d_kernel(const float* __restrict__ s, u16* __restrict__ d,
                      int sr, int sc, int dr, int dc)
{
  const int total = dr * dc;
  const int stride = gridDim.x * blockDim.x;
  for (int i = blockIdx.x * blockDim.x + threadIdx.x; i < total; i += stride) {
    const int r = i / dc;
    const int c = i % dc;
    d[i] = (r < sr && c < sc) ? f2bf_bits(s[r * sc + c]) : (u16)0;
  }
}

// dst[np] (f32) = src[n] zero-padded
__global__ __launch_bounds__(64)
void pad_f32_kernel(const float* __restrict__ s, float* __restrict__ d, int n, int np)
{
  const int i = blockIdx.x * blockDim.x + threadIdx.x;
  if (i < np) d[i] = (i < n) ? s[i] : 0.0f;
}

__global__ void softmax5_kernel(const float* __restrict__ a, float* __restrict__ o)
{
  if (threadIdx.x == 0 && blockIdx.x == 0) {
    float m = a[0];
    for (int i = 1; i < 5; ++i) m = fmaxf(m, a[i]);
    float e[5], s = 0.0f;
    for (int i = 0; i < 5; ++i) { e[i] = __expf(a[i] - m); s += e[i]; }
    for (int i = 0; i < 5; ++i) o[i] = e[i] / s;
  }
}

__global__ __launch_bounds__(256)
void pos_write_kernel(float* __restrict__ out, const float* __restrict__ x6, int n)
{
  const int stride = gridDim.x * blockDim.x;
  for (int i = blockIdx.x * blockDim.x + threadIdx.x; i < n; i += stride)
    out[i] = 0.5f * x6[i];
}

__global__ __launch_bounds__(256)
void neg_combine_kernel(float* __restrict__ out,
                        const float* __restrict__ x3L, const float* __restrict__ x3,
                        const float* __restrict__ x4,  const float* __restrict__ x5,
                        const float* __restrict__ x6,  const float* __restrict__ a, int n)
{
  const float a0 = a[0], a1 = a[1], a2 = a[2], a3 = a[3], a4 = a[4];
  const int stride = gridDim.x * blockDim.x;
  for (int i = blockIdx.x * blockDim.x + threadIdx.x; i < n; i += stride)
    out[i] -= 0.5f * (a0 * x3L[i] + a1 * x3[i] + a2 * x4[i] + a3 * x5[i] + a4 * x6[i]);
}

// ---------------------------------------------------------------------------
extern "C" void kernel_launch(void* const* d_in, const int* in_sizes, int n_in,
                              void* d_out, int out_size, void* d_ws, size_t ws_size,
                              hipStream_t stream)
{
  const int NN = 4096, DIN = 512;
  (void)in_sizes; (void)n_in; (void)out_size; (void)ws_size;

  const float* A_pos = (const float*)d_in[1];
  const float* A_neg = (const float*)d_in[2];
  const float* attn  = (const float*)d_in[39];
  float* out = (float*)d_out;

  // ---- workspace bump allocator (256B aligned -> all b128 loads aligned) ----
  char* wsb = (char*)d_ws;
  size_t off = 0;
  auto alloc = [&](size_t bytes) -> char* {
    char* p = wsb + off;
    off = (off + bytes + 255) & ~(size_t)255;
    return p;
  };
  u16*   anB   = (u16*)alloc((size_t)NN * NN * 2);     // 32 MB, reused per branch
  float* dis   = (float*)alloc((size_t)NN * 4);
  u16*   featB = (u16*)alloc((size_t)NN * DIN * 2);
  u16*   wl1 = (u16*)alloc((size_t)512 * 256 * 2);
  u16*   wl2 = (u16*)alloc((size_t)256 * 64 * 2);      // 256x62 -> 256x64
  u16*   wl3 = (u16*)alloc((size_t)64 * 128 * 2);      // 62x128 -> 64x128
  u16*   wg1 = (u16*)alloc((size_t)256 * 256 * 2);
  u16*   wg2 = (u16*)alloc((size_t)64 * 64 * 2);       // 62x62 -> 64x64
  u16*   wg3 = (u16*)alloc((size_t)128 * 128 * 2);
  u16*   wg4 = (u16*)alloc((size_t)128 * 128 * 2);
  u16*   wg5 = (u16*)alloc((size_t)128 * 128 * 2);
  u16*   wg6 = (u16*)alloc((size_t)128 * 128 * 2);
  float* bl2p = (float*)alloc(64 * 4);                 // padded biases (62 -> 64)
  float* bg2p = (float*)alloc(64 * 4);
  u16*   Pb  = (u16*)alloc((size_t)NN * 256 * 2);
  float* x1Lf = (float*)alloc((size_t)NN * 256 * 4);
  u16*   x1Lb = (u16*)alloc((size_t)NN * 256 * 2);
  float* x1f  = (float*)alloc((size_t)NN * 256 * 4);
  u16*   x1b  = (u16*)alloc((size_t)NN * 256 * 2);
  float* x2Lf = (float*)alloc((size_t)NN * 64 * 4);    // 62 -> 64 padded
  u16*   x2Lb = (u16*)alloc((size_t)NN * 64 * 2);
  float* x2f  = (float*)alloc((size_t)NN * 64 * 4);
  u16*   x2b  = (u16*)alloc((size_t)NN * 64 * 2);
  float* x3Lf = (float*)alloc((size_t)NN * 128 * 4);
  u16*   x3Lb = (u16*)alloc((size_t)NN * 128 * 2);
  float* x3f  = (float*)alloc((size_t)NN * 128 * 4);
  u16*   x3b  = (u16*)alloc((size_t)NN * 128 * 2);
  float* x4f  = (float*)alloc((size_t)NN * 128 * 4);
  u16*   x4b  = (u16*)alloc((size_t)NN * 128 * 2);
  float* x5f  = (float*)alloc((size_t)NN * 128 * 4);
  u16*   x5b  = (u16*)alloc((size_t)NN * 128 * 2);
  float* x6f  = (float*)alloc((size_t)NN * 128 * 4);
  float* asoft = (float*)alloc(8 * 4);

  // ---- launch helpers ----
  auto cvt2d = [&](const void* s, u16* d, int sr, int sc, int dr, int dc) {
    int n = dr * dc;
    int b = (n + 255) / 256; if (b > 4096) b = 4096;
    cvt_pad2d_kernel<<<b, 256, 0, stream>>>((const float*)s, d, sr, sc, dr, dc);
  };
  auto gemm = [&](const u16* A, const u16* B, const float* bias, const float* res,
                  float* oF, u16* oB, int M, int N, int K, float scale, int mode) {
    dim3 g(N / BN, M / BM);
    switch (mode) {
      case MODE_RAW:
        gemm_bf16_wmma<MODE_RAW><<<g, 256, 0, stream>>>(A, B, bias, res, oF, oB, M, N, K, scale); break;
      case MODE_BIAS:
        gemm_bf16_wmma<MODE_BIAS><<<g, 256, 0, stream>>>(A, B, bias, res, oF, oB, M, N, K, scale); break;
      case MODE_BIAS_RELU:
        gemm_bf16_wmma<MODE_BIAS_RELU><<<g, 256, 0, stream>>>(A, B, bias, res, oF, oB, M, N, K, scale); break;
      case MODE_RES_RELU:
        gemm_bf16_wmma<MODE_RES_RELU><<<g, 256, 0, stream>>>(A, B, bias, res, oF, oB, M, N, K, scale); break;
      default:
        gemm_bf16_wmma<MODE_RES_LIN><<<g, 256, 0, stream>>>(A, B, bias, res, oF, oB, M, N, K, scale); break;
    }
  };

  cvt2d(d_in[0], featB, NN, DIN, NN, DIN);   // feature -> bf16 (once)

  // Common front of each branch: normalize A, project, GCN1/2 residual blocks,
  // then x3L = x2 @ lin3 + b.   (62-wide tensors padded to 64 with exact zeros)
  auto run_front = [&](const float* Aadj, int base) {
    rowsum_rsqrt_kernel<<<NN, 256, 0, stream>>>(Aadj, dis, NN);
    build_an_kernel<<<4096, 256, 0, stream>>>(Aadj, dis, anB, NN);
    cvt2d(d_in[base + 0],  wl1, 512, 256, 512, 256);
    cvt2d(d_in[base + 2],  wl2, 256, 62, 256, 64);
    cvt2d(d_in[base + 4],  wl3, 62, 128, 64, 128);
    cvt2d(d_in[base + 6],  wg1, 256, 256, 256, 256);
    cvt2d(d_in[base + 8],  wg2, 62, 62, 64, 64);
    cvt2d(d_in[base + 10], wg3, 128, 128, 128, 128);
    cvt2d(d_in[base + 12], wg4, 128, 128, 128, 128);
    cvt2d(d_in[base + 14], wg5, 128, 128, 128, 128);
    cvt2d(d_in[base + 16], wg6, 128, 128, 128, 128);
    pad_f32_kernel<<<1, 64, 0, stream>>>((const float*)d_in[base + 3], bl2p, 62, 64);
    pad_f32_kernel<<<1, 64, 0, stream>>>((const float*)d_in[base + 9], bg2p, 62, 64);
    const float* b_l1 = (const float*)d_in[base + 1];
    const float* b_l3 = (const float*)d_in[base + 5];
    const float* b_g1 = (const float*)d_in[base + 7];

    gemm(featB, wl1, b_l1, nullptr, x1Lf, x1Lb, NN, 256, DIN, 1.0f, MODE_BIAS);
    gemm(x1Lb, wg1, nullptr, nullptr, nullptr, Pb, NN, 256, 256, 1.0f, MODE_RAW);
    gemm(anB, Pb, b_g1, x1Lf, x1f, x1b, NN, 256, NN, 1.0f, MODE_RES_RELU);

    gemm(x1b, wl2, bl2p, nullptr, x2Lf, x2Lb, NN, 64, 256, 1.0f, MODE_BIAS);
    gemm(x2Lb, wg2, nullptr, nullptr, nullptr, Pb, NN, 64, 64, 1.0f, MODE_RAW);
    gemm(anB, Pb, bg2p, x2Lf, x2f, x2b, NN, 64, NN, 1.0f, MODE_RES_RELU);

    gemm(x2b, wl3, b_l3, nullptr, x3Lf, x3Lb, NN, 128, 64, 1.0f, MODE_BIAS);
  };

  // ================= positive branch =================
  {
    const int base = 3;
    run_front(A_pos, base);
    const float* b_g3 = (const float*)d_in[base + 11];
    const float* b_g4 = (const float*)d_in[base + 13];
    const float* b_g5 = (const float*)d_in[base + 15];
    const float* b_g6 = (const float*)d_in[base + 17];

    gemm(x3Lb, wg3, nullptr, nullptr, nullptr, Pb, NN, 128, 128, 1.0f, MODE_RAW);
    gemm(anB, Pb, b_g3, x3Lf, x3f, x3b, NN, 128, NN, 0.5f, MODE_RES_RELU);

    gemm(x3b, wg4, nullptr, nullptr, nullptr, Pb, NN, 128, 128, 1.0f, MODE_RAW);
    gemm(anB, Pb, b_g4, x3f, x4f, x4b, NN, 128, NN, 0.5f, MODE_RES_RELU);

    gemm(x4b, wg5, nullptr, nullptr, nullptr, Pb, NN, 128, 128, 1.0f, MODE_RAW);
    gemm(anB, Pb, b_g5, x4f, x5f, x5b, NN, 128, NN, 0.25f, MODE_RES_RELU);

    gemm(x5b, wg6, nullptr, nullptr, nullptr, Pb, NN, 128, 128, 1.0f, MODE_RAW);
    gemm(anB, Pb, b_g6, x5f, x6f, nullptr, NN, 128, NN, 0.25f, MODE_RES_LIN);

    pos_write_kernel<<<2048, 256, 0, stream>>>(out, x6f, NN * 128);
  }

  // ================= negative branch =================
  {
    const int base = 21;
    run_front(A_neg, base);
    const float* b_g3 = (const float*)d_in[base + 11];
    const float* b_g4 = (const float*)d_in[base + 13];
    const float* b_g5 = (const float*)d_in[base + 15];
    const float* b_g6 = (const float*)d_in[base + 17];

    gemm(x3Lb, wg3, nullptr, nullptr, nullptr, Pb, NN, 128, 128, 1.0f, MODE_RAW);
    gemm(anB, Pb, b_g3, nullptr, x3f, x3b, NN, 128, NN, 1.0f, MODE_BIAS_RELU);

    gemm(x3b, wg4, nullptr, nullptr, nullptr, Pb, NN, 128, 128, 1.0f, MODE_RAW);
    gemm(anB, Pb, b_g4, nullptr, x4f, x4b, NN, 128, NN, 1.0f, MODE_BIAS_RELU);

    gemm(x4b, wg5, nullptr, nullptr, nullptr, Pb, NN, 128, 128, 1.0f, MODE_RAW);
    gemm(anB, Pb, b_g5, nullptr, x5f, x5b, NN, 128, NN, 1.0f, MODE_BIAS_RELU);

    gemm(x5b, wg6, nullptr, nullptr, nullptr, Pb, NN, 128, 128, 1.0f, MODE_RAW);
    gemm(anB, Pb, b_g6, nullptr, x6f, nullptr, NN, 128, NN, 1.0f, MODE_BIAS);

    softmax5_kernel<<<1, 32, 0, stream>>>(attn, asoft);
    neg_combine_kernel<<<2048, 256, 0, stream>>>(out, x3Lf, x3f, x4f, x5f, x6f, asoft, NN * 128);
  }
}
